// Block_867583394304
// MI455X (gfx1250) — matrix operand order repaired and checked by reading, not compiled, not run on hardware
//
#include <hip/hip_runtime.h>
#include <hip/hip_bf16.h>

typedef __attribute__((ext_vector_type(16))) __bf16 bf16x16;
typedef __attribute__((ext_vector_type(8)))  float  f32x8;
typedef unsigned short ushort_t;

// ---------------- helpers ----------------

__device__ __forceinline__ ushort_t f2bf(float f) {
  union { float f; unsigned u; } c; c.f = f;
  unsigned u = c.u;
  u += 0x7fffu + ((u >> 16) & 1u);   // round-to-nearest-even
  return (ushort_t)(u >> 16);
}

__device__ __forceinline__ float wave_sum(float v) {
#pragma unroll
  for (int o = 16; o > 0; o >>= 1) v += __shfl_xor(v, o, 32);
  return v;
}
__device__ __forceinline__ float wave_max(float v) {
#pragma unroll
  for (int o = 16; o > 0; o >>= 1) v = fmaxf(v, __shfl_xor(v, o, 32));
  return v;
}

// ---------------- elementwise prep ----------------

__global__ void f32_to_bf16_kernel(const float* __restrict__ in,
                                   ushort_t* __restrict__ out, int n) {
  int i = blockIdx.x * 256 + threadIdx.x;
  if (i < n) out[i] = f2bf(in[i]);
}

__global__ void build_qkv_bias_kernel(const float* __restrict__ qb,
                                      const float* __restrict__ vb,
                                      float* __restrict__ out) {
  int i = blockIdx.x * 256 + threadIdx.x;
  if (i < 2304) out[i] = (i < 768) ? qb[i] : (i < 1536 ? 0.f : vb[i - 1536]);
}

// ---------------- LayerNorm (row of 768) -> bf16 ----------------

__global__ __launch_bounds__(256) void ln_bf16_kernel(
    const float* __restrict__ x, const float* __restrict__ w,
    const float* __restrict__ b, ushort_t* __restrict__ out) {
  const int row = blockIdx.x;
  const int tid = threadIdx.x;
  const float* xr = x + (size_t)row * 768;
  float v[3];
  float s = 0.f, s2 = 0.f;
#pragma unroll
  for (int i = 0; i < 3; ++i) {
    v[i] = xr[tid + 256 * i];
    s += v[i]; s2 += v[i] * v[i];
  }
  __shared__ float red[16];
  s = wave_sum(s); s2 = wave_sum(s2);
  const int wid = tid >> 5, lane = tid & 31;
  if (lane == 0) { red[wid] = s; red[8 + wid] = s2; }
  __syncthreads();
  if (wid == 0) {
    float a  = (lane < 8) ? red[lane] : 0.f;
    float a2 = (lane < 8) ? red[8 + lane] : 0.f;
    a = wave_sum(a); a2 = wave_sum(a2);
    if (lane == 0) { red[0] = a; red[1] = a2; }
  }
  __syncthreads();
  const float mean = red[0] * (1.f / 768.f);
  const float var  = red[1] * (1.f / 768.f) - mean * mean;
  const float inv  = rsqrtf(var + 1e-5f);
#pragma unroll
  for (int i = 0; i < 3; ++i) {
    int c = tid + 256 * i;
    out[(size_t)row * 768 + c] = f2bf((v[i] - mean) * inv * w[c] + b[c]);
  }
}

// ---------------- WMMA GEMM: C[m,n] = sum_k A[m,k] * W[n,k] (+epilogue) ----
// A: [M,K] bf16 row-major, W: [N,K] bf16 row-major. K must be a multiple of 64.
// Block tile 128x128, 8 waves as 4x2, each wave 32x64 = 2x4 wmma tiles, BK=32.
// Double-buffered LDS with statically-named buffers; K-loop processes two
// 32-steps per iteration -> one barrier per step, branch-free body.
// mode 0: out_bf16 = acc + bias
// mode 1: out_bf16 = gelu(acc + bias)          (exact, erf)
// mode 2: out_f32  = resid + gamma * (acc + bias)

#define GLDT 40  // LDS pitch in bf16 elems (64B row + 16B pad)

__global__ __launch_bounds__(256) void gemm_wmma_kernel(
    const ushort_t* __restrict__ A, const ushort_t* __restrict__ W,
    const float* __restrict__ bias, const float* __restrict__ gamma,
    const float* __restrict__ resid, ushort_t* __restrict__ outb,
    float* __restrict__ outf, int M, int N, int K, int mode) {
  __shared__ ushort_t sA0[128 * GLDT];
  __shared__ ushort_t sA1[128 * GLDT];
  __shared__ ushort_t sB0[128 * GLDT];
  __shared__ ushort_t sB1[128 * GLDT];

  const int tid = threadIdx.x;
  const int mBase = blockIdx.y * 128;
  const int nBase = blockIdx.x * 128;
  const int wv = tid >> 5, lane = tid & 31;
  const int wr = wv >> 1, wc = wv & 1;   // wave grid 4 (M) x 2 (N)
  const int lr = lane & 15, g = lane >> 4;

  // this thread's two staging chunks (row, 16B-chunk-in-row)
  const int c0row = tid >> 2,          c0c4 = tid & 3;
  const int c1row = (tid + 256) >> 2,  c1c4 = tid & 3;
  const ushort_t* gA0 = A + (size_t)(mBase + c0row) * K + c0c4 * 8;
  const ushort_t* gA1 = A + (size_t)(mBase + c1row) * K + c1c4 * 8;
  const ushort_t* gW0 = W + (size_t)(nBase + c0row) * K + c0c4 * 8;
  const ushort_t* gW1 = W + (size_t)(nBase + c1row) * K + c1c4 * 8;
  const int s0 = c0row * GLDT + c0c4 * 8;
  const int s1 = c1row * GLDT + c1c4 * 8;

  f32x8 acc[2][4];
#pragma unroll
  for (int i = 0; i < 2; ++i)
#pragma unroll
    for (int j = 0; j < 4; ++j)
#pragma unroll
      for (int e = 0; e < 8; ++e) acc[i][j][e] = 0.f;

  auto stage = [&](ushort_t* dA, ushort_t* dB, int k) {
    *(uint4*)(dA + s0) = *(const uint4*)(gA0 + k);
    *(uint4*)(dA + s1) = *(const uint4*)(gA1 + k);
    *(uint4*)(dB + s0) = *(const uint4*)(gW0 + k);
    *(uint4*)(dB + s1) = *(const uint4*)(gW1 + k);
  };

  auto compute = [&](const ushort_t* bA, const ushort_t* bB) {
    union Fr { uint4 q[2]; bf16x16 v; } fa[2], fb[4];
#pragma unroll
    for (int mt = 0; mt < 2; ++mt) {
      const ushort_t* p = bA + (wr * 32 + mt * 16 + lr) * GLDT;
      fa[mt].q[0] = *(const uint4*)(p + 8 * g);        // K = 8g..8g+7
      fa[mt].q[1] = *(const uint4*)(p + 16 + 8 * g);   // K = 16+8g..
    }
#pragma unroll
    for (int nt = 0; nt < 4; ++nt) {
      const ushort_t* p = bB + (wc * 64 + nt * 16 + lr) * GLDT;
      fb[nt].q[0] = *(const uint4*)(p + 8 * g);        // K = 8g..8g+7
      fb[nt].q[1] = *(const uint4*)(p + 16 + 8 * g);   // K = 16+8g..
    }
#pragma unroll
    for (int mt = 0; mt < 2; ++mt)
#pragma unroll
      for (int nt = 0; nt < 4; ++nt)
        acc[mt][nt] = __builtin_amdgcn_wmma_f32_16x16x32_bf16(
            false, fa[mt].v, false, fb[nt].v, (short)0, acc[mt][nt],
            false, false);
  };

  stage(sA0, sB0, 0);  // prologue
  for (int k0 = 0; k0 < K; k0 += 64) {
    __syncthreads();                         // sA0/sB0 (k0) visible
    stage(sA1, sB1, k0 + 32);                // k0+32 <= K-32, always valid
    {                                        // warm next lines (global_prefetch_b8)
      const int kp = (k0 + 96 < K) ? k0 + 96 : 0;
      __builtin_prefetch(gA0 + kp, 0, 3);
      __builtin_prefetch(gW0 + kp, 0, 3);
    }
    compute(sA0, sB0);
    __syncthreads();                         // sA1/sB1 (k0+32) visible
    const int kn = (k0 + 64 < K) ? k0 + 64 : 0;  // last iter: harmless restage
    stage(sA0, sB0, kn);
    compute(sA1, sB1);
  }

#pragma unroll
  for (int mt = 0; mt < 2; ++mt) {
#pragma unroll
    for (int nt = 0; nt < 4; ++nt) {
      const int n = nBase + wc * 64 + nt * 16 + lr;
      const float bs = bias ? bias[n] : 0.f;
      const float gm = (mode == 2) ? gamma[n] : 0.f;
#pragma unroll
      for (int r = 0; r < 8; ++r) {
        const int m = mBase + wr * 32 + mt * 16 + g * 8 + r;
        float v = acc[mt][nt][r] + bs;
        if (mode == 1) v = 0.5f * v * (1.f + erff(v * 0.70710678118f));
        if (mode == 2) {
          outf[(size_t)m * N + n] = resid[(size_t)m * N + n] + gm * v;
        } else {
          outb[(size_t)m * N + n] = f2bf(v);
        }
      }
    }
  }
}

// ---------------- attention: one block = (b, h, 16-query tile) -------------
// qkv: [8192, 2304] bf16, token row = [q(768)|k(768)|v(768)], head h at h*64.
// out: [8192, 768] bf16.

#define SP 1028   // S fp32 pitch (breaks 8-row bank alias: 8*1028 % 64 == 32)
#define PP 1032   // P bf16 pitch (16B aligned)
#define VP 40     // Vt bf16 pitch

#define ATTN_SMEM (16 * SP * 4 + 16 * PP * 2 + 2 * 64 * VP * 2)

__global__ __launch_bounds__(128) void attn_kernel(
    const ushort_t* __restrict__ qkv, ushort_t* __restrict__ out) {
  extern __shared__ char smem[];
  float*    S   = (float*)smem;
  ushort_t* P   = (ushort_t*)(smem + 16 * SP * 4);
  ushort_t* Vt0 = (ushort_t*)(smem + 16 * SP * 4 + 16 * PP * 2);
  ushort_t* Vt1 = Vt0 + 64 * VP;

  const int blk = blockIdx.x;         // b*768 + h*64 + qi
  const int qi  = blk & 63;
  const int h   = (blk >> 6) % 12;
  const int b   = blk / (12 * 64);
  const int tid = threadIdx.x;
  const int wv = tid >> 5, lane = tid & 31;
  const int lr = lane & 15, g = lane >> 4;

  const size_t rs = 2304;
  const size_t base = (size_t)(b * 1024) * rs;

  union Fr { uint4 q[2]; bf16x16 v; };

  // q fragments for this 16-row query tile (K = 64 -> two K=32 chunks)
  Fr qf[2];
  {
    const ushort_t* qrow = qkv + base + (size_t)(qi * 16 + lr) * rs + h * 64;
#pragma unroll
    for (int c = 0; c < 2; ++c) {
      qf[c].q[0] = *(const uint4*)(qrow + c * 32 + 8 * g);
      qf[c].q[1] = *(const uint4*)(qrow + c * 32 + 16 + 8 * g);
    }
  }

  // ---- S = (q k^T) * scale ; waves stripe the 64 key tiles ----
  const float scale = 0.125f;  // 1/sqrt(64)
  for (int jt = wv; jt < 64; jt += 4) {
    f32x8 sacc;
#pragma unroll
    for (int e = 0; e < 8; ++e) sacc[e] = 0.f;
    const ushort_t* krow = qkv + base + (size_t)(jt * 16 + lr) * rs + 768 + h * 64;
#pragma unroll
    for (int c = 0; c < 2; ++c) {
      Fr kf;
      kf.q[0] = *(const uint4*)(krow + c * 32 + 8 * g);
      kf.q[1] = *(const uint4*)(krow + c * 32 + 16 + 8 * g);
      sacc = __builtin_amdgcn_wmma_f32_16x16x32_bf16(
          false, qf[c].v, false, kf.v, (short)0, sacc, false, false);
    }
#pragma unroll
    for (int r = 0; r < 8; ++r)
      S[(r + 8 * g) * SP + jt * 16 + lr] = sacc[r] * scale;
  }
  __syncthreads();

  // ---- softmax over 1024 keys; 4 rows per wave ----
#pragma unroll
  for (int rr = 0; rr < 4; ++rr) {
    const int row = wv * 4 + rr;
    float m = -1e30f;
    for (int c = 0; c < 32; ++c) m = fmaxf(m, S[row * SP + c * 32 + lane]);
    m = wave_max(m);
    float s = 0.f;
    for (int c = 0; c < 32; ++c) {
      float e = __expf(S[row * SP + c * 32 + lane] - m);
      S[row * SP + c * 32 + lane] = e;
      s += e;
    }
    s = wave_sum(s);
    const float inv = 1.f / s;
    for (int c = 0; c < 32; ++c)
      P[row * PP + c * 32 + lane] = f2bf(S[row * SP + c * 32 + lane] * inv);
  }

  // ---- O = P V ; each wave owns one 16-wide d-tile, K stepped by 32.
  //      V-transpose staging double-buffered (static buffers, two steps
  //      per iteration, branch-free): one barrier per step. ----
  const ushort_t* vbase = qkv + base + 1536 + h * 64;
  const int sjj = tid >> 6;           // staging coords: 16 (jj,d) pairs
  const int sd  = tid & 63;           //   jj = sjj + 2*i
  f32x8 oacc;
#pragma unroll
  for (int e = 0; e < 8; ++e) oacc[e] = 0.f;

  auto stageV = [&](ushort_t* dst, int j) {
#pragma unroll
    for (int i = 0; i < 16; ++i) {
      int jj = sjj + 2 * i;
      dst[sd * VP + jj] = vbase[(size_t)(j + jj) * rs + sd];
    }
  };
  auto computePV = [&](const ushort_t* vt, int j) {
    Fr pf, vf;
    const ushort_t* prow = P + lr * PP + j;
    pf.q[0] = *(const uint4*)(prow + 8 * g);
    pf.q[1] = *(const uint4*)(prow + 16 + 8 * g);
    const ushort_t* vrow = vt + (wv * 16 + lr) * VP;
    vf.q[0] = *(const uint4*)(vrow + 8 * g);
    vf.q[1] = *(const uint4*)(vrow + 16 + 8 * g);
    oacc = __builtin_amdgcn_wmma_f32_16x16x32_bf16(
        false, pf.v, false, vf.v, (short)0, oacc, false, false);
  };

  stageV(Vt0, 0);  // prologue (P written above; first barrier covers both)
  for (int j0 = 0; j0 < 1024; j0 += 64) {
    __syncthreads();                       // Vt0 (j0) + P visible
    stageV(Vt1, j0 + 32);                  // j0+32 <= 992, always valid
    computePV(Vt0, j0);
    __syncthreads();                       // Vt1 (j0+32) visible
    const int jn = (j0 + 64 < 1024) ? j0 + 64 : 0;  // last iter: harmless
    stageV(Vt0, jn);
    computePV(Vt1, j0 + 32);
  }
#pragma unroll
  for (int r = 0; r < 8; ++r) {
    const int q = qi * 16 + r + 8 * g;
    out[(size_t)(b * 1024 + q) * 768 + h * 64 + wv * 16 + lr] = f2bf(oacc[r]);
  }
}

// ---------------- launch ----------------

extern "C" void kernel_launch(void* const* d_in, const int* in_sizes, int n_in,
                              void* d_out, int out_size, void* d_ws,
                              size_t ws_size, hipStream_t stream) {
  (void)in_sizes; (void)n_in; (void)out_size; (void)ws_size;
  const float* x      = (const float*)d_in[0];
  const float* ln1_w  = (const float*)d_in[1];
  const float* ln1_b  = (const float*)d_in[2];
  const float* qkv_w  = (const float*)d_in[3];
  const float* q_bias = (const float*)d_in[4];
  const float* v_bias = (const float*)d_in[5];
  const float* proj_w = (const float*)d_in[6];
  const float* proj_b = (const float*)d_in[7];
  const float* ln2_w  = (const float*)d_in[8];
  const float* ln2_b  = (const float*)d_in[9];
  const float* fc1_w  = (const float*)d_in[10];
  const float* fc1_b  = (const float*)d_in[11];
  const float* fc2_w  = (const float*)d_in[12];
  const float* fc2_b  = (const float*)d_in[13];
  const float* gamma1 = (const float*)d_in[14];
  const float* gamma2 = (const float*)d_in[15];

  char* ws = (char*)d_ws;
  size_t off = 0;
  auto alloc = [&](size_t bytes) -> void* {
    void* p = ws + off;
    off += (bytes + 255) & ~(size_t)255;
    return p;
  };
  const int M = 8192;  // 8 * 1024 tokens

  ushort_t* qkv_wb  = (ushort_t*)alloc((size_t)2304 * 768 * 2);
  ushort_t* proj_wb = (ushort_t*)alloc((size_t)768 * 768 * 2);
  ushort_t* fc1_wb  = (ushort_t*)alloc((size_t)3072 * 768 * 2);
  ushort_t* fc2_wb  = (ushort_t*)alloc((size_t)768 * 3072 * 2);
  float*    qkv_bias= (float*)alloc(2304 * 4);
  ushort_t* xn      = (ushort_t*)alloc((size_t)M * 768 * 2);
  ushort_t* qkvb    = (ushort_t*)alloc((size_t)M * 2304 * 2);
  ushort_t* attn_o  = (ushort_t*)alloc((size_t)M * 768 * 2);
  float*    h1      = (float*)alloc((size_t)M * 768 * 4);
  ushort_t* xn2     = (ushort_t*)alloc((size_t)M * 768 * 2);
  ushort_t* hmlp    = (ushort_t*)alloc((size_t)M * 3072 * 2);

  // weights -> bf16
  f32_to_bf16_kernel<<<(2304 * 768 + 255) / 256, 256, 0, stream>>>(qkv_w, qkv_wb, 2304 * 768);
  f32_to_bf16_kernel<<<(768 * 768 + 255) / 256, 256, 0, stream>>>(proj_w, proj_wb, 768 * 768);
  f32_to_bf16_kernel<<<(3072 * 768 + 255) / 256, 256, 0, stream>>>(fc1_w, fc1_wb, 3072 * 768);
  f32_to_bf16_kernel<<<(768 * 3072 + 255) / 256, 256, 0, stream>>>(fc2_w, fc2_wb, 768 * 3072);
  build_qkv_bias_kernel<<<9, 256, 0, stream>>>(q_bias, v_bias, qkv_bias);

  // ln1 -> bf16
  ln_bf16_kernel<<<M, 256, 0, stream>>>(x, ln1_w, ln1_b, xn);

  // qkv = xn @ qkv_w^T + qkv_bias  -> bf16 [M, 2304]
  gemm_wmma_kernel<<<dim3(18, 64), 256, 0, stream>>>(
      xn, qkv_wb, qkv_bias, nullptr, nullptr, qkvb, nullptr, M, 2304, 768, 0);

  // attention -> bf16 [M, 768]
  hipFuncSetAttribute((const void*)attn_kernel,
                      hipFuncAttributeMaxDynamicSharedMemorySize, ATTN_SMEM);
  attn_kernel<<<6144, 128, ATTN_SMEM, stream>>>(qkvb, attn_o);

  // h1 = x + gamma1 * (attn_o @ proj_w^T + proj_b)  -> fp32 [M, 768]
  gemm_wmma_kernel<<<dim3(6, 64), 256, 0, stream>>>(
      attn_o, proj_wb, proj_b, gamma1, x, nullptr, h1, M, 768, 768, 2);

  // ln2 -> bf16
  ln_bf16_kernel<<<M, 256, 0, stream>>>(h1, ln2_w, ln2_b, xn2);

  // hmlp = gelu(xn2 @ fc1_w^T + fc1_b)  -> bf16 [M, 3072]
  gemm_wmma_kernel<<<dim3(24, 64), 256, 0, stream>>>(
      xn2, fc1_wb, fc1_b, nullptr, nullptr, hmlp, nullptr, M, 3072, 768, 1);

  // out = h1 + gamma2 * (hmlp @ fc2_w^T + fc2_b)  -> fp32 d_out
  gemm_wmma_kernel<<<dim3(6, 64), 256, 0, stream>>>(
      hmlp, fc2_wb, fc2_b, gamma2, h1, nullptr, (float*)d_out, M, 768, 3072, 2);
}
